// DCRNN_Temporal_30459908063367
// MI455X (gfx1250) — matrix operand order repaired and checked by reading, not compiled, not run on hardware
//
#include <hip/hip_runtime.h>
#include <hip/hip_bf16.h>

#define NN 50000
#define NE 800000
#define FIN 32
#define FOUT 64
#define NP 12
#define KTOT 160   // [x | T1o | T1i | T2o | T2i] -> 5 chunks of 32

typedef __attribute__((ext_vector_type(16))) _Float16 v16h;
typedef __attribute__((ext_vector_type(8)))  float    v8f;

// ---------------- init: zero degrees, zero T1, T2 = -x, M chunk0 = f16(x) ---
__global__ void init_kernel(const float* __restrict__ x,
                            float* __restrict__ degO, float* __restrict__ degI,
                            float* __restrict__ T1o, float* __restrict__ T1i,
                            float* __restrict__ T2o, float* __restrict__ T2i,
                            _Float16* __restrict__ M) {
  int i = blockIdx.x * blockDim.x + threadIdx.x;
  if (i < NN * FIN) {
    float xv = x[i];
    T1o[i] = 0.f; T1i[i] = 0.f;
    T2o[i] = -xv; T2i[i] = -xv;            // T2 = 2*prop(T1) - x
    int n = i >> 5, f = i & 31;
    M[(size_t)n * KTOT + f] = (_Float16)xv;
  }
  if (i < NN) { degO[i] = 0.f; degI[i] = 0.f; }
}

// ---------------- weighted degrees -----------------------------------------
__global__ void deg_kernel(const int* __restrict__ ei, const float* __restrict__ ew,
                           float* __restrict__ degO, float* __restrict__ degI) {
  int e = blockIdx.x * blockDim.x + threadIdx.x;
  if (e >= NE) return;
  float w = ew[e];
  atomicAdd(&degO[ei[e]], w);        // row
  atomicAdd(&degI[ei[NE + e]], w);   // col
}

__global__ void inv_kernel(float* __restrict__ degO, float* __restrict__ degI) {
  int n = blockIdx.x * blockDim.x + threadIdx.x;
  if (n >= NN) return;
  float a = degO[n]; degO[n] = (a != 0.f) ? 1.f / a : 0.f;
  float b = degI[n]; degI[n] = (b != 0.f) ? 1.f / b : 0.f;
}

// ---------------- edge propagation (both directions, float4 per thread) ----
__global__ void prop_kernel(const int* __restrict__ ei,
                            const float* __restrict__ invO, const float* __restrict__ invI,
                            const float* __restrict__ srcF, const float* __restrict__ srcR,
                            float* __restrict__ dstF, float* __restrict__ dstR,
                            float scale) {
  int idx = blockIdx.x * blockDim.x + threadIdx.x;
  if (idx >= NE * 8) return;
  int e = idx >> 3, f4 = idx & 7;
  int r = ei[e], c = ei[NE + e];
  float no = scale * invO[r];
  float ni = scale * invI[c];
  float4 sF = ((const float4*)srcF)[r * 8 + f4];   // gather row of srcF
  float4 sR = ((const float4*)srcR)[c * 8 + f4];   // gather col of srcR
  float* dF = dstF + (size_t)c * FIN + f4 * 4;
  float* dR = dstR + (size_t)r * FIN + f4 * 4;
  atomicAdd(dF + 0, no * sF.x); atomicAdd(dF + 1, no * sF.y);
  atomicAdd(dF + 2, no * sF.z); atomicAdd(dF + 3, no * sF.w);
  atomicAdd(dR + 0, ni * sR.x); atomicAdd(dR + 1, ni * sR.y);
  atomicAdd(dR + 2, ni * sR.z); atomicAdd(dR + 3, ni * sR.w);
}

// ---------------- pack two f32 feature blocks into f16 feature matrix ------
__global__ void pack_kernel(const float* __restrict__ A, const float* __restrict__ B,
                            _Float16* __restrict__ M, int offA, int offB) {
  int i = blockIdx.x * blockDim.x + threadIdx.x;
  if (i >= NN * FIN) return;
  int n = i >> 5, f = i & 31;
  M[(size_t)n * KTOT + offA + f] = (_Float16)A[i];
  M[(size_t)n * KTOT + offB + f] = (_Float16)B[i];
}

// ---------------- pack gate weights: Wt[g][o][k] f16 (column-major K x N) --
__global__ void packw_kernel(const float* __restrict__ wz, const float* __restrict__ wh,
                             _Float16* __restrict__ Wt) {
  int idx = blockIdx.x * blockDim.x + threadIdx.x;
  if (idx >= 2 * FOUT * KTOT) return;
  int g = idx / (FOUT * KTOT);
  int rem = idx - g * (FOUT * KTOT);
  int o = rem / KTOT;
  int k = rem - o * KTOT;
  const float* w = g ? wh : wz;   // (2,3,96,64): idx ((d*3+kk)*96+c)*64+o
  float val;
  if (k < 32) {                                     // w[0,0] + w[1,0]
    val = w[(0 * 96 + k) * 64 + o] + w[(3 * 96 + k) * 64 + o];
  } else {
    int seg = (k - 32) >> 5;                        // 0:T1o 1:T1i 2:T2o 3:T2i
    int c   = (k - 32) & 31;
    int d   = seg & 1;
    int kk  = 1 + (seg >> 1);
    val = w[((d * 3 + kk) * 96 + c) * 64 + o];
  }
  Wt[idx] = (_Float16)val;                          // idx == (g*64+o)*160+k
}

// ---------------- WMMA GEMM + gate epilogue + linear head ------------------
__global__ __launch_bounds__(256)
void gemm_kernel(const _Float16* __restrict__ M, const _Float16* __restrict__ Wt,
                 const float* __restrict__ bz, const float* __restrict__ bh,
                 const float* __restrict__ wlin, const float* __restrict__ blin,
                 float* __restrict__ out) {
  __shared__ float Hs[128 * FOUT];
  __shared__ float wl[FOUT * NP];
  __shared__ float bl[NP];
  int tid = threadIdx.x;
  for (int i = tid; i < FOUT * NP; i += 256) wl[i] = wlin[i];
  if (tid < NP) bl[tid] = blin[tid];

  int wave = tid >> 5, lane = tid & 31;
  int node0 = blockIdx.x * 128;
  int mrow = node0 + wave * 16 + (lane & 15);
  if (mrow > NN - 1) mrow = NN - 1;                 // clamp; stores are guarded
  const _Float16* Arow = M + (size_t)mrow * KTOT;
  int kbA = (lane < 16) ? 0 : 16;                   // 16-bit A 16x32 layout (contiguous k = 16h + s)
  int kbB = (lane < 16) ? 0 : 16;                   // 16-bit B 32x16 layout
  int ncol = lane & 15;

  v8f accz[4] = {};
  v8f acch[4] = {};

  #pragma unroll
  for (int kc = 0; kc < 5; ++kc) {
    v16h a;
    const _Float16* pa = Arow + kc * 32;
    #pragma unroll
    for (int j = 0; j < 8; ++j) {                   // VGPR j holds K = kk,kk+1
      int kk = kbA + 2 * j;
      a[2 * j]     = pa[kk];
      a[2 * j + 1] = pa[kk + 1];
    }
    #pragma unroll
    for (int t = 0; t < 4; ++t) {                   // 4 output tiles of 16 cols
      const _Float16* qz = Wt + (size_t)(t * 16 + ncol) * KTOT + kc * 32 + kbB;
      v16h bzf = *(const v16h*)qz;
      accz[t] = __builtin_amdgcn_wmma_f32_16x16x32_f16(
          false, a, false, bzf, (short)0, accz[t], false, false);
      const _Float16* qh = Wt + (size_t)(FOUT + t * 16 + ncol) * KTOT + kc * 32 + kbB;
      v16h bhf = *(const v16h*)qh;
      acch[t] = __builtin_amdgcn_wmma_f32_16x16x32_f16(
          false, a, false, bhf, (short)0, acch[t], false, false);
    }
  }

  // epilogue: Z=sigmoid, Ht=tanh, H=(1-Z)*Ht, ReLU -> LDS
  #pragma unroll
  for (int t = 0; t < 4; ++t) {
    int n = t * 16 + ncol;
    float bzn = bz[n], bhn = bh[n];
    #pragma unroll
    for (int r = 0; r < 8; ++r) {                   // C/D: VGPR r -> M = r | 8+r
      int ml = wave * 16 + ((lane < 16) ? r : 8 + r);
      float z  = 1.f / (1.f + __expf(-(accz[t][r] + bzn)));
      float ht = tanhf(acch[t][r] + bhn);
      float h  = (1.f - z) * ht;
      Hs[ml * FOUT + n] = h > 0.f ? h : 0.f;
    }
  }
  __syncthreads();

  // head: relu(H) @ w_lin + b_lin
  for (int idx = tid; idx < 128 * NP; idx += 256) {
    int ln = idx / NP, p = idx - ln * NP;
    int node = node0 + ln;
    if (node < NN) {
      float s = bl[p];
      const float* hr = Hs + ln * FOUT;
      #pragma unroll
      for (int f = 0; f < FOUT; ++f) s += hr[f] * wl[f * NP + p];
      out[(size_t)node * NP + p] = s;
    }
  }
}

extern "C" void kernel_launch(void* const* d_in, const int* in_sizes, int n_in,
                              void* d_out, int out_size, void* d_ws, size_t ws_size,
                              hipStream_t stream) {
  const float* x    = (const float*)d_in[0];
  const int*   ei   = (const int*)d_in[1];
  const float* ew   = (const float*)d_in[2];
  const float* w_z  = (const float*)d_in[3];
  const float* b_z  = (const float*)d_in[4];
  // d_in[5], d_in[6] = w_r, b_r: R gate multiplies H0==0 -> provably unused
  const float* w_h  = (const float*)d_in[7];
  const float* b_h  = (const float*)d_in[8];
  const float* wlin = (const float*)d_in[9];
  const float* blin = (const float*)d_in[10];
  float* out = (float*)d_out;

  char* ws = (char*)d_ws;
  size_t off = 0;
  auto alloc = [&](size_t bytes) -> char* {
    char* r = ws + off;
    off = (off + bytes + 511) & ~(size_t)511;
    return r;
  };
  float* degO = (float*)alloc((size_t)NN * 4);
  float* degI = (float*)alloc((size_t)NN * 4);
  float* T1o  = (float*)alloc((size_t)NN * FIN * 4);
  float* T1i  = (float*)alloc((size_t)NN * FIN * 4);
  float* T2o  = (float*)alloc((size_t)NN * FIN * 4);
  float* T2i  = (float*)alloc((size_t)NN * FIN * 4);
  _Float16* M  = (_Float16*)alloc((size_t)NN * KTOT * 2);
  _Float16* Wt = (_Float16*)alloc((size_t)2 * FOUT * KTOT * 2);

  const int B = 256;
  init_kernel<<<(NN * FIN + B - 1) / B, B, 0, stream>>>(x, degO, degI, T1o, T1i, T2o, T2i, M);
  deg_kernel<<<(NE + B - 1) / B, B, 0, stream>>>(ei, ew, degO, degI);
  inv_kernel<<<(NN + B - 1) / B, B, 0, stream>>>(degO, degI);
  packw_kernel<<<(2 * FOUT * KTOT + B - 1) / B, B, 0, stream>>>(w_z, w_h, Wt);
  // T1 = P_fwd x, P_rev x
  prop_kernel<<<(NE * 8 + B - 1) / B, B, 0, stream>>>(ei, degO, degI, x, x, T1o, T1i, 1.0f);
  pack_kernel<<<(NN * FIN + B - 1) / B, B, 0, stream>>>(T1o, T1i, M, 32, 64);
  // T2 = 2*P(T1) - x  (dst pre-initialized to -x)
  prop_kernel<<<(NE * 8 + B - 1) / B, B, 0, stream>>>(ei, degO, degI, T1o, T1i, T2o, T2i, 2.0f);
  pack_kernel<<<(NN * FIN + B - 1) / B, B, 0, stream>>>(T2o, T2i, M, 96, 128);
  gemm_kernel<<<(NN + 127) / 128, B, 0, stream>>>(M, Wt, b_z, b_h, wlin, blin, out);
}
